// MoeLayerTopPLayer_63496796504066
// MI455X (gfx1250) — matrix operand order repaired and checked by reading, not compiled, not run on hardware
//
#include <hip/hip_runtime.h>
#include <hip/hip_bf16.h>
#include <math.h>

typedef __attribute__((ext_vector_type(16))) __bf16 v16bf;
typedef __attribute__((ext_vector_type(8)))  __bf16 v8bf;
typedef __attribute__((ext_vector_type(8)))  float  v8f;

#define B_ 8192
#define D_ 1024
#define H_ 2048
#define O_ 1024
#define E_ 8
#define MT 32          // rows per workgroup (2x 16-row WMMA halves)
#define NTH 512        // 16 waves
#define TOPP 0.9f

// ---------- helpers ----------
__device__ __forceinline__ __bf16 f2bf(float f) {
  unsigned u = __builtin_bit_cast(unsigned, f);
  unsigned r = u + 0x7FFFu + ((u >> 16) & 1u);       // round-to-nearest-even
  unsigned short h = (unsigned short)(r >> 16);
  return __builtin_bit_cast(__bf16, h);
}

// A-fragment (16x32 bf16) from a row-major bf16 tile in LDS.
// Lanes 0-15 hold M=0..15 with K [0,16); lanes 16-31 get K [16,32) (matches B swizzle).
__device__ __forceinline__ v16bf load_afrag(const __bf16* base, int ldk, int mrow,
                                            int k0, int halfSel) {
  const __bf16* r = base + mrow * ldk + k0 + halfSel * 16;
  v8bf r0 = *(const v8bf*)(r);
  v8bf r1 = *(const v8bf*)(r + 8);
  return __builtin_shufflevector(r0, r1, 0,1,2,3,4,5,6,7,8,9,10,11,12,13,14,15);
}

// B-fragment (32x16 bf16): weights pre-swizzled so each lane's 16 values are
// contiguous (32B, fully coalesced 1KB/wave). blk = (expert,ktile,ntile) block.
__device__ __forceinline__ v16bf load_bfrag(const __bf16* w, long long blk, int lane) {
  const v16bf* p = (const v16bf*)(w + (size_t)blk * 512);
  return p[lane];
}

// ---------- conversion kernels ----------
__global__ void cvt_x(const float* __restrict__ x, __bf16* __restrict__ xb) {
  size_t i = (size_t)blockIdx.x * blockDim.x + threadIdx.x;
  if (i < (size_t)B_ * D_) xb[i] = f2bf(x[i]);
}

__global__ void cvt_w_swizzle(const float* __restrict__ w, __bf16* __restrict__ out,
                              int K, int N) {
  size_t i = (size_t)blockIdx.x * blockDim.x + threadIdx.x;
  size_t total = (size_t)E_ * K * N;
  if (i >= total) return;
  int n = (int)(i % N);
  size_t t = i / N;
  int k = (int)(t % K);
  int e = (int)(t / K);
  int kt = k >> 5, kr = k & 31, nt = n >> 4, nr = n & 15;
  int lane = ((kr >> 4) << 4) | nr;
  int elem = kr & 15;
  size_t off = (((size_t)e * (K >> 5) + kt) * (N >> 4) + nt) * 512 + (size_t)lane * 16 + elem;
  out[off] = f2bf(w[i]);
}

// ---------- gate: logits, softmax, entropy, top-p weights ----------
__global__ __launch_bounds__(256) void gate_kernel(const float* __restrict__ x,
    const float* __restrict__ gw, const float* __restrict__ gb,
    float* __restrict__ wts, float* __restrict__ psum, float* __restrict__ rsum) {
  int wave = threadIdx.x >> 5, lane = threadIdx.x & 31;
  int row = blockIdx.x * 8 + wave;
  if (row >= B_) return;
  float acc[E_];
#pragma unroll
  for (int e = 0; e < E_; ++e) acc[e] = 0.f;
  const float* xr = x + (size_t)row * D_;
  for (int d = lane; d < D_; d += 32) {
    float xv = xr[d];
    const float* g = gw + d * E_;
#pragma unroll
    for (int e = 0; e < E_; ++e) acc[e] = fmaf(xv, g[e], acc[e]);
  }
#pragma unroll
  for (int e = 0; e < E_; ++e)
#pragma unroll
    for (int off = 16; off > 0; off >>= 1)
      acc[e] += __shfl_xor(acc[e], off, 32);
  if (lane == 0) {
    float p[E_], mx = -1e30f;
#pragma unroll
    for (int e = 0; e < E_; ++e) { p[e] = acc[e] + gb[e]; mx = fmaxf(mx, p[e]); }
    float s = 0.f;
#pragma unroll
    for (int e = 0; e < E_; ++e) { p[e] = __expf(p[e] - mx); s += p[e]; }
    float inv = 1.f / s, ent = 0.f;
#pragma unroll
    for (int e = 0; e < E_; ++e) { p[e] *= inv; ent -= p[e] * logf(p[e] + 1e-8f); }
    int order[E_]; bool used[E_];
#pragma unroll
    for (int e = 0; e < E_; ++e) used[e] = false;
    for (int i = 0; i < E_; ++i) {
      int best = -1; float bv = -1.f;
      for (int e = 0; e < E_; ++e)
        if (!used[e] && p[e] > bv) { bv = p[e]; best = e; }
      used[best] = true; order[i] = best;
    }
    float cum = 0.f; float mp[E_];
    for (int i = 0; i < E_; ++i) {
      cum += p[order[i]];
      bool m = (cum <= TOPP) || (i == 0);
      mp[order[i]] = m ? p[order[i]] : 0.f;
    }
    float ms = 0.f;
#pragma unroll
    for (int e = 0; e < E_; ++e) ms += mp[e];
    float minv = 1.f / (ms + 1e-8f);
    float* wr = wts + (size_t)row * E_;
#pragma unroll
    for (int e = 0; e < E_; ++e) wr[e] = mp[e] * minv;
    atomicAdd(rsum, ent);
#pragma unroll
    for (int e = 0; e < E_; ++e) atomicAdd(&psum[e], p[e]);
  }
}

// ---------- fused MoE: 32-row tile, loop experts, fuse gemm1+gelu+gemm2 ----------
__global__ __launch_bounds__(NTH) void moe_fused(
    const __bf16* __restrict__ xb, const __bf16* __restrict__ w1s,
    const __bf16* __restrict__ w2s, const float* __restrict__ b1,
    const float* __restrict__ b2, const float* __restrict__ wts,
    float* __restrict__ esum, float* __restrict__ outp) {
  __shared__ __align__(16) __bf16 sX[MT][D_];   // 64 KB
  __shared__ __align__(16) __bf16 sH[MT][512];  // 32 KB
  __shared__ float s_wts[MT];

  const int tid = threadIdx.x;
  const int wave = tid >> 5, lane = tid & 31;      // 16 waves
  const int halfSel = lane >> 4;
  const int mrow = lane & 15, ncol = lane & 15;
  const int rowBase = blockIdx.x * MT;

  // async global->LDS staging of the 32x1024 x-tile (64KB, 16B per op, ASYNCcnt)
  for (int i = tid; i < MT * (D_ / 8); i += NTH) {  // 4096 16B chunks
    int r = i >> 7, c = i & 127;
    const __bf16* g = xb + (size_t)(rowBase + r) * D_ + c * 8;
    unsigned l = (unsigned)(uintptr_t)(&sX[r][c * 8]);
    asm volatile("global_load_async_to_lds_b128 %0, %1, off"
                 :: "v"(l), "v"(g) : "memory");
  }
  asm volatile("s_wait_asynccnt 0" ::: "memory");
  __syncthreads();

  v8f acc_out[2][4];
#pragma unroll
  for (int mh = 0; mh < 2; ++mh)
#pragma unroll
    for (int n = 0; n < 4; ++n)
#pragma unroll
      for (int j = 0; j < 8; ++j) acc_out[mh][n][j] = 0.f;

  for (int e = 0; e < E_; ++e) {
    __syncthreads();
    if (tid < MT) s_wts[tid] = wts[(size_t)(rowBase + tid) * E_ + e];
    __syncthreads();

    v8f acc2[2][4];  // [row-half][ntile]: 32 rows x 64 cols of out_e per wave
#pragma unroll
    for (int n = 0; n < 4; ++n) {
      float bv = b2[e * O_ + wave * 64 + n * 16 + ncol];
#pragma unroll
      for (int mh = 0; mh < 2; ++mh)
#pragma unroll
        for (int j = 0; j < 8; ++j) acc2[mh][n][j] = bv;
    }

    for (int hc = 0; hc < 4; ++hc) {  // H in chunks of 512
      // ---- phase 1: h_chunk = gelu(x @ w1[e][:, chunk] + b1); 32 cols/wave ----
      v8f acc1[2][2];
#pragma unroll
      for (int n = 0; n < 2; ++n) {
        float bv = b1[e * H_ + hc * 512 + wave * 32 + n * 16 + ncol];
#pragma unroll
        for (int mh = 0; mh < 2; ++mh)
#pragma unroll
          for (int j = 0; j < 8; ++j) acc1[mh][n][j] = bv;
      }
      const int nb1 = hc * 32 + wave * 2;  // H-tile base (of 128)
      for (int k = 0; k < D_; k += 32) {
        int kt = k >> 5;
        v16bf a0 = load_afrag(&sX[0][0],  D_, mrow, k, halfSel);
        v16bf a1 = load_afrag(&sX[16][0], D_, mrow, k, halfSel);
        if (k + 32 < D_)
          __builtin_prefetch(w1s + (((size_t)(e * 32 + kt + 1) * 128 + nb1) * 512), 0, 3);
#pragma unroll
        for (int n = 0; n < 2; ++n) {
          v16bf b = load_bfrag(w1s, (long long)(e * 32 + kt) * 128 + nb1 + n, lane);
          acc1[0][n] = __builtin_amdgcn_wmma_f32_16x16x32_bf16(
              false, a0, false, b, (short)0, acc1[0][n], false, false);
          acc1[1][n] = __builtin_amdgcn_wmma_f32_16x16x32_bf16(
              false, a1, false, b, (short)0, acc1[1][n], false, false);
        }
      }
      // exact-erf GELU -> bf16 LDS
#pragma unroll
      for (int mh = 0; mh < 2; ++mh)
#pragma unroll
        for (int n = 0; n < 2; ++n)
#pragma unroll
          for (int j = 0; j < 8; ++j) {
            float v = acc1[mh][n][j];
            float g = 0.5f * v * (1.f + erff(v * 0.70710678118654752f));
            int m = mh * 16 + (halfSel ? (j + 8) : j);
            sH[m][wave * 32 + n * 16 + ncol] = f2bf(g);
          }
      __syncthreads();

      // ---- phase 2: acc2 += h_chunk @ w2[e][chunk, :]; 64 cols/wave ----
      for (int kk = 0; kk < 512; kk += 32) {
        v16bf a0 = load_afrag(&sH[0][0],  512, mrow, kk, halfSel);
        v16bf a1 = load_afrag(&sH[16][0], 512, mrow, kk, halfSel);
        int kt = (hc * 512 + kk) >> 5;
#pragma unroll
        for (int n = 0; n < 4; ++n) {
          v16bf b = load_bfrag(w2s, (long long)(e * 64 + kt) * 64 + wave * 4 + n, lane);
          acc2[0][n] = __builtin_amdgcn_wmma_f32_16x16x32_bf16(
              false, a0, false, b, (short)0, acc2[0][n], false, false);
          acc2[1][n] = __builtin_amdgcn_wmma_f32_16x16x32_bf16(
              false, a1, false, b, (short)0, acc2[1][n], false, false);
        }
      }
      __syncthreads();  // sH reused next chunk
    }

    // weighted combine into persistent output + expert column sums
#pragma unroll
    for (int n = 0; n < 4; ++n) {
      float cs = 0.f;
#pragma unroll
      for (int mh = 0; mh < 2; ++mh)
#pragma unroll
        for (int j = 0; j < 8; ++j) {
          float o = acc2[mh][n][j];
          float wr = s_wts[mh * 16 + (halfSel ? (j + 8) : j)];
          acc_out[mh][n][j] = fmaf(wr, o, acc_out[mh][n][j]);
          cs += o;
        }
      cs += __shfl_xor(cs, 16, 32);  // combine both lane-halves of the column
      if (halfSel == 0)
        atomicAdd(&esum[e * O_ + wave * 64 + n * 16 + ncol], cs);
    }
  }

  // store output tile
#pragma unroll
  for (int mh = 0; mh < 2; ++mh)
#pragma unroll
    for (int n = 0; n < 4; ++n)
#pragma unroll
      for (int j = 0; j < 8; ++j) {
        int row = rowBase + mh * 16 + (halfSel ? (j + 8) : j);
        outp[(size_t)row * O_ + wave * 64 + n * 16 + ncol] = acc_out[mh][n][j];
      }
}

// ---------- losses: entropy mean + gram logdet via Cholesky ----------
__global__ __launch_bounds__(256) void finalize_kernel(
    const float* __restrict__ psum, const float* __restrict__ rsum,
    const float* __restrict__ esum, float* __restrict__ out) {
  __shared__ float sV[E_][O_];
  __shared__ float red[256];
  __shared__ float snorm[E_];
  __shared__ float gram[E_ * E_];
  int tid = threadIdx.x;
  for (int i = tid; i < E_ * O_; i += 256) sV[i / O_][i % O_] = esum[i] * (1.f / B_);
  __syncthreads();
  for (int e = 0; e < E_; ++e) {
    float s = 0.f;
    for (int c = tid; c < O_; c += 256) { float v = sV[e][c]; s += v * v; }
    red[tid] = s; __syncthreads();
    for (int st = 128; st > 0; st >>= 1) {
      if (tid < st) red[tid] += red[tid + st];
      __syncthreads();
    }
    if (tid == 0) snorm[e] = sqrtf(red[0]) + 1e-6f;
    __syncthreads();
  }
  for (int i = tid; i < E_ * O_; i += 256) {
    int e = i / O_;
    sV[e][i % O_] *= (psum[e] * (1.f / B_)) / snorm[e];
  }
  __syncthreads();
  if (tid < 64) {
    int i = tid >> 3, j = tid & 7;
    float s = 0.f;
    for (int c = 0; c < O_; ++c) s += sV[i][c] * sV[j][c];
    if (i == j) s += 1e-6f;
    gram[tid] = s;
  }
  __syncthreads();
  if (tid == 0) {
    float L[E_][E_];
    float logdet = 0.f;
    for (int k = 0; k < E_; ++k) {
      float s = gram[k * E_ + k];
      for (int m = 0; m < k; ++m) s -= L[k][m] * L[k][m];
      float d = sqrtf(fmaxf(s, 1e-30f));
      L[k][k] = d; logdet += 2.f * logf(d);
      for (int i2 = k + 1; i2 < E_; ++i2) {
        float t = gram[i2 * E_ + k];
        for (int m = 0; m < k; ++m) t -= L[i2][m] * L[k][m];
        L[i2][k] = t / d;
      }
    }
    out[(size_t)B_ * O_]     = rsum[0] * (1.f / B_);
    out[(size_t)B_ * O_ + 1] = -logdet;
  }
}

// ---------- workspace layout (bytes) ----------
#define WS_XB   ((size_t)0)                        // B*D bf16      = 16 MB
#define WS_W1   ((size_t)16777216)                 // E*D*H bf16    = 32 MB
#define WS_W2   ((size_t)50331648)                 // E*H*O bf16    = 32 MB
#define WS_WTS  ((size_t)83886080)                 // B*E f32       = 256 KB
#define WS_ACC  ((size_t)84148224)                 // psum(8f)|rsum@+32|esum@+64

extern "C" void kernel_launch(void* const* d_in, const int* in_sizes, int n_in,
                              void* d_out, int out_size, void* d_ws, size_t ws_size,
                              hipStream_t stream) {
  const float* x   = (const float*)d_in[0];
  const float* gw  = (const float*)d_in[1];
  const float* gb  = (const float*)d_in[2];
  const float* w1  = (const float*)d_in[3];
  const float* b1  = (const float*)d_in[4];
  const float* w2  = (const float*)d_in[5];
  const float* b2  = (const float*)d_in[6];
  float* out = (float*)d_out;
  char* ws = (char*)d_ws;

  __bf16* xb  = (__bf16*)(ws + WS_XB);
  __bf16* w1s = (__bf16*)(ws + WS_W1);
  __bf16* w2s = (__bf16*)(ws + WS_W2);
  float*  wts = (float*)(ws + WS_WTS);
  float*  psum = (float*)(ws + WS_ACC);
  float*  rsum = (float*)(ws + WS_ACC + 32);
  float*  esum = (float*)(ws + WS_ACC + 64);

  hipMemsetAsync(ws + WS_ACC, 0, 64 + (size_t)E_ * O_ * sizeof(float), stream);

  cvt_x<<<(B_ * D_) / 256, 256, 0, stream>>>(x, xb);
  cvt_w_swizzle<<<((size_t)E_ * D_ * H_) / 256, 256, 0, stream>>>(w1, w1s, D_, H_);
  cvt_w_swizzle<<<((size_t)E_ * H_ * O_) / 256, 256, 0, stream>>>(w2, w2s, H_, O_);

  gate_kernel<<<B_ / 8, 256, 0, stream>>>(x, gw, gb, wts, psum, rsum);

  moe_fused<<<B_ / MT, NTH, 0, stream>>>(xb, w1s, w2s, b1, b2, wts, esum, out);

  finalize_kernel<<<1, 256, 0, stream>>>(psum, rsum, esum, out);
}
